// LSTMModel_19688130085847
// MI455X (gfx1250) — matrix operand order repaired and checked by reading, not compiled, not run on hardware
//
#include <hip/hip_runtime.h>

// ---------------------------------------------------------------------------
// Fused 2-layer LSTM (H=50, T=512, B=4096, I=1) + final FC, for gfx1250.
//
// Per workgroup (128 thr = 4 waves): one batch tile of 16 rows.
// Wave w owns hidden units [16w,16w+16) of BOTH layers, computing the
// i/f/g/o gate tiles for those units with v_wmma_f32_16x16x32_f16.
// Gate layout: N = gate*64 + unit (each gate padded 50->64) so the four
// gate accumulators for a unit share lanes -> register-local cell update.
// K layout (layer1): k<50 = h1, k=50 = x_t (row = w_ih1), k=51 = 1.0
// (row = b_ih1+b_hh1).  Layer2 K=128: k<50 = h1(t), k=51 -> b2, k in
// [64,114) = h2(t-1).  Weight B-fragments are held in VGPRs (built once);
// h-state ping-pongs in LDS (f16), A-fragments = 2x ds_load_b128.
// ---------------------------------------------------------------------------

typedef __attribute__((ext_vector_type(16))) _Float16 v16h;
typedef __attribute__((ext_vector_type(8)))  _Float16 v8h;
typedef __attribute__((ext_vector_type(8)))  float    v8f;

#define HID 50
#define SEQ 512
#define KP  64      // padded K per hidden vector

__device__ __forceinline__ float fast_exp2(float v) {
#if defined(__AMDGCN__) && __has_builtin(__builtin_amdgcn_exp2f)
  return __builtin_amdgcn_exp2f(v);           // v_exp_f32
#else
  return exp2f(v);
#endif
}

__device__ __forceinline__ float tanh_fast(float v) {
#if defined(__gfx1250__) && __has_builtin(__builtin_amdgcn_tanhf)
  return __builtin_amdgcn_tanhf(v);           // v_tanh_f32 (CDNA5 TRANS op)
#else
  float ax = fminf(fabsf(v), 10.0f);
  float e  = fast_exp2(2.8853900817779268f * ax);   // exp(2*ax)
  float tt = 1.0f - 2.0f / (e + 1.0f);
  return copysignf(tt, v);
#endif
}

__device__ __forceinline__ float sigmoid_fast(float v) {
  return 0.5f * tanh_fast(0.5f * v) + 0.5f;   // saturates safely, no clamp
}

// Layer-1 augmented weight element: W1[k][n], n = gate*64 + j, k in [0,64)
__device__ __forceinline__ float w1_elem(int j, int gate, int k,
                                         const float* __restrict__ w_hh1,
                                         const float* __restrict__ w_ih1,
                                         const float* __restrict__ b_ih1,
                                         const float* __restrict__ b_hh1) {
  if (j >= HID) return 0.0f;
  int row = gate * HID + j;                   // PyTorch gate order i,f,g,o
  if (k < HID) return w_hh1[row * HID + k];
  if (k == 50) return w_ih1[row];             // x slot (w_ih1 is [200,1])
  if (k == 51) return b_ih1[row] + b_hh1[row];// bias slot (*1.0)
  return 0.0f;
}

// Layer-2 augmented weight element: k in [0,128): [0,64) = h1 side, [64,128) = h2
__device__ __forceinline__ float w2_elem(int j, int gate, int k,
                                         const float* __restrict__ w_ih2,
                                         const float* __restrict__ w_hh2,
                                         const float* __restrict__ b_ih2,
                                         const float* __restrict__ b_hh2) {
  if (j >= HID) return 0.0f;
  int row = gate * HID + j;
  if (k < 64) {
    if (k < HID) return w_ih2[row * HID + k];
    if (k == 51) return b_ih2[row] + b_hh2[row];  // 1.0 lives at col 51
    return 0.0f;                                  // k==50 is x_t: ignored
  }
  int kk = k - 64;
  if (kk < HID) return w_hh2[row * HID + kk];
  return 0.0f;
}

__device__ __forceinline__ v8f wmma16(v16h a, v16h b, v8f c) {
  return __builtin_amdgcn_wmma_f32_16x16x32_f16(false, a, false, b,
                                                (short)0, c, false, false);
}

// A fragment (16x32 f16): lane = m + 16*kh holds K runs [8kh,8kh+8) and
// [16+8kh, 16+8kh+8) -> two contiguous ds_load_b128 per lane.
__device__ __forceinline__ v16h load_afrag(const _Float16* buf, int m, int kh,
                                           int ks) {
  const _Float16* p = buf + m * KP + ks * 32 + kh * 8;
  v8h lo = *(const v8h*)p;
  v8h hi = *(const v8h*)(p + 16);
  v16h a;
#pragma unroll
  for (int e = 0; e < 8; ++e) { a[e] = lo[e]; a[e + 8] = hi[e]; }
  return a;
}

__global__ __launch_bounds__(128)
void lstm2_fused(const float* __restrict__ x,
                 const float* __restrict__ w_ih1, const float* __restrict__ w_hh1,
                 const float* __restrict__ b_ih1, const float* __restrict__ b_hh1,
                 const float* __restrict__ w_ih2, const float* __restrict__ w_hh2,
                 const float* __restrict__ b_ih2, const float* __restrict__ b_hh2,
                 const float* __restrict__ fc_w, const float* __restrict__ fc_b,
                 float* __restrict__ out) {
  __shared__ __align__(16) _Float16 h1ds[2][16 * KP];  // ping-pong h1 (+x,+1.0)
  __shared__ __align__(16) _Float16 h2ds[2][16 * KP];  // ping-pong h2

  const int tid  = threadIdx.x;
  const int lane = tid & 31;
  const int si   = tid >> 5;        // this wave's hidden sub-block (0..3)
  const int nloc = lane & 15;       // n within tile / batch row m for A
  const int kh   = lane >> 4;       // lane half
  const int b0   = blockIdx.x * 16; // batch tile base
  const int j    = si * 16 + nloc;  // hidden unit this lane produces

  // ---- build this wave's weight B-fragments, resident in VGPRs ----
  // B (32x16 f16): lane = n + 16*kh holds K runs [k0+8kh,+8) and [k0+16+8kh,+8).
  v16h w1B[4][2];                   // [gate][kstep]   (layer 1, K=64)
  v16h w2B[4][4];                   // [gate][kstep]   (layer 2, K=128)
#pragma unroll
  for (int g = 0; g < 4; ++g) {
#pragma unroll
    for (int ks = 0; ks < 2; ++ks)
#pragma unroll
      for (int e = 0; e < 16; ++e) {
        int k = ks * 32 + (e < 8 ? kh * 8 + e : 16 + kh * 8 + (e - 8));
        w1B[g][ks][e] = (_Float16)w1_elem(j, g, k, w_hh1, w_ih1, b_ih1, b_hh1);
      }
#pragma unroll
    for (int ks = 0; ks < 4; ++ks)
#pragma unroll
      for (int e = 0; e < 16; ++e) {
        int k = ks * 32 + (e < 8 ? kh * 8 + e : 16 + kh * 8 + (e - 8));
        w2B[g][ks][e] = (_Float16)w2_elem(j, g, k, w_ih2, w_hh2, b_ih2, b_hh2);
      }
  }

  // ---- init LDS state: zeros, 1.0 at col 51 (bias hook), x_0 at col 50 ----
  for (int i = tid; i < 16 * KP; i += blockDim.x) {
    h1ds[0][i] = (_Float16)0.0f; h1ds[1][i] = (_Float16)0.0f;
    h2ds[0][i] = (_Float16)0.0f; h2ds[1][i] = (_Float16)0.0f;
  }
  __syncthreads();
  if (tid < 16) {
    h1ds[0][tid * KP + 51] = (_Float16)1.0f;
    h1ds[1][tid * KP + 51] = (_Float16)1.0f;
    h1ds[0][tid * KP + 50] = (_Float16)x[(size_t)(b0 + tid) * SEQ + 0];
  }

  const v8f zero8 = {0.f, 0.f, 0.f, 0.f, 0.f, 0.f, 0.f, 0.f};
  v8f c1 = zero8, c2 = zero8;

  for (int t = 0; t < SEQ; ++t) {
    const int cur = t & 1, nxt = cur ^ 1;
    __syncthreads();                               // B0: h(t-1)+x_t visible

    v16h a1k0 = load_afrag(h1ds[cur], nloc, kh, 0);
    v16h a1k1 = load_afrag(h1ds[cur], nloc, kh, 1);
    v16h a2k0 = load_afrag(h2ds[cur], nloc, kh, 0);
    v16h a2k1 = load_afrag(h2ds[cur], nloc, kh, 1);

    // ---- layer 1: gates = [h1|x|1] x W1  (8 WMMAs) ----
    v8f acc1[4];
#pragma unroll
    for (int g = 0; g < 4; ++g) {
      v8f a = zero8;
      a = wmma16(a1k0, w1B[g][0], a);
      a = wmma16(a1k1, w1B[g][1], a);
      acc1[g] = a;
    }
    v8f h1v = zero8;
#pragma unroll
    for (int r = 0; r < 8; ++r) {
      float iv = sigmoid_fast(acc1[0][r]);
      float fv = sigmoid_fast(acc1[1][r]);
      float gv = tanh_fast(acc1[2][r]);
      float ov = sigmoid_fast(acc1[3][r]);
      float c  = fv * c1[r] + iv * gv;
      c1[r]    = c;
      h1v[r]   = ov * tanh_fast(c);
    }
    if (j < HID) {                                 // write h1(t) (cols 50+ are x/1/pad)
#pragma unroll
      for (int r = 0; r < 8; ++r)
        h1ds[nxt][(r + kh * 8) * KP + j] = (_Float16)h1v[r];
    }
    if (tid < 16 && t + 1 < SEQ)                   // stage x_{t+1} (wave 0)
      h1ds[nxt][tid * KP + 50] = (_Float16)x[(size_t)(b0 + tid) * SEQ + (t + 1)];

    __syncthreads();                               // B1: h1(t) visible

    v16h a1n0 = load_afrag(h1ds[nxt], nloc, kh, 0);
    v16h a1n1 = load_afrag(h1ds[nxt], nloc, kh, 1);

    // ---- layer 2: gates = [h1(t)|1 , h2(t-1)] x W2  (16 WMMAs) ----
    v8f acc2[4];
#pragma unroll
    for (int g = 0; g < 4; ++g) {
      v8f a = zero8;
      a = wmma16(a1n0, w2B[g][0], a);
      a = wmma16(a1n1, w2B[g][1], a);
      a = wmma16(a2k0, w2B[g][2], a);
      a = wmma16(a2k1, w2B[g][3], a);
      acc2[g] = a;
    }
    v8f h2v = zero8;
#pragma unroll
    for (int r = 0; r < 8; ++r) {
      float iv = sigmoid_fast(acc2[0][r]);
      float fv = sigmoid_fast(acc2[1][r]);
      float gv = tanh_fast(acc2[2][r]);
      float ov = sigmoid_fast(acc2[3][r]);
      float c  = fv * c2[r] + iv * gv;
      c2[r]    = c;
      h2v[r]   = ov * tanh_fast(c);
    }
    if (j < HID) {
#pragma unroll
      for (int r = 0; r < 8; ++r)
        h2ds[nxt][(r + kh * 8) * KP + j] = (_Float16)h2v[r];
    }
  }

  __syncthreads();
  // ---- final FC: out[b] = h2_T . fc_w + fc_b  (SEQ even -> buffer 0) ----
  if (tid < 16) {
    const _Float16* hrow = &h2ds[SEQ & 1][tid * KP];
    float acc = fc_b[0];
#pragma unroll 10
    for (int jj = 0; jj < HID; ++jj) acc += (float)hrow[jj] * fc_w[jj];
    out[b0 + tid] = acc;
  }
}

extern "C" void kernel_launch(void* const* d_in, const int* in_sizes, int n_in,
                              void* d_out, int out_size, void* d_ws, size_t ws_size,
                              hipStream_t stream) {
  (void)n_in; (void)d_ws; (void)ws_size;
  const float* x     = (const float*)d_in[0];
  const float* w_ih1 = (const float*)d_in[1];
  const float* w_hh1 = (const float*)d_in[2];
  const float* b_ih1 = (const float*)d_in[3];
  const float* b_hh1 = (const float*)d_in[4];
  const float* w_ih2 = (const float*)d_in[5];
  const float* w_hh2 = (const float*)d_in[6];
  const float* b_ih2 = (const float*)d_in[7];
  const float* b_hh2 = (const float*)d_in[8];
  const float* fc_w  = (const float*)d_in[9];
  const float* fc_b  = (const float*)d_in[10];

  const int B = in_sizes[0] / SEQ;      // 4096
  (void)out_size;
  dim3 grid(B / 16), block(128);
  lstm2_fused<<<grid, block, 0, stream>>>(x, w_ih1, w_hh1, b_ih1, b_hh1,
                                          w_ih2, w_hh2, b_ih2, b_hh2,
                                          fc_w, fc_b, (float*)d_out);
}
